// Attention_1056561955116
// MI455X (gfx1250) — compile-verified, matrix operand order checked by reading, not run on hardware
//
#include <hip/hip_runtime.h>
#include <hip/hip_bf16.h>
#include <math.h>

#define BATCH 2
#define SEQ   2048
#define DIMX  1024
#define HEADS 16
#define HDIM  64
#define CHUNKSZ 128
#define NCHK  (SEQ / CHUNKSZ)       // 16
#define INNER (HEADS * HDIM)        // 1024
#define BSROWS (BATCH * SEQ)        // 4096

typedef __attribute__((ext_vector_type(8)))  float  v8f;
typedef __attribute__((ext_vector_type(16))) __bf16 v16bf;

union Frag { v16bf v; uint4 q[2]; };

__device__ __forceinline__ unsigned short f2bf(float f) {
  unsigned int u = __float_as_uint(f);
  return (unsigned short)((u + 0x7FFFu + ((u >> 16) & 1u)) >> 16);   // RNE
}
__device__ __forceinline__ float bf2f(unsigned short h) {
  return __uint_as_float(((unsigned int)h) << 16);
}
__device__ __forceinline__ v8f wmma_bf16(const Frag& a, const Frag& b, v8f c) {
  return __builtin_amdgcn_wmma_f32_16x16x32_bf16(false, a.v, false, b.v,
                                                 (short)0, c, false, false);
}

// ---------------- Tensor Data Mover: 2D bf16 tile, global -> LDS ----------------
#if defined(__gfx1250__) && __has_builtin(__builtin_amdgcn_tensor_load_to_lds) && \
    __has_builtin(__builtin_amdgcn_s_wait_tensorcnt)
#define USE_TDM 1
#else
#define USE_TDM 0
#endif

#if USE_TDM
typedef unsigned int u32x4 __attribute__((ext_vector_type(4)));
typedef int i32x4 __attribute__((ext_vector_type(4)));
typedef int i32x8 __attribute__((ext_vector_type(8)));

// D# (groups 0/1) for a 2D tile of bf16: tile_w elems/row, tile_h rows,
// source row stride src_stride elems.  tensor == tile -> no OOB clipping.
__device__ __forceinline__ void tdm_load_2d_bf16(const unsigned short* lds_dst,
                                                 const unsigned short* gsrc,
                                                 unsigned tile_w, unsigned tile_h,
                                                 unsigned src_stride) {
  unsigned long long ga = (unsigned long long)gsrc;
  unsigned lds = (unsigned)(unsigned long long)lds_dst;  // flat LDS addr[31:0] = LDS offset
  u32x4 g0;
  g0.x = 1u;                                  // count=1, is_restore=0, gather off
  g0.y = lds;                                 // lds_addr  (bits 63:32)
  g0.z = (unsigned)ga;                        // global_addr[31:0]   (bits 95:64)
  g0.w = (unsigned)(ga >> 32) | (2u << 30);   // global_addr[56:32] | type=2
  i32x8 g1;
  g1[0] = (int)(1u << 16);                          // wg_mask=0, data_size=1 (2B)
  g1[1] = (int)((tile_w & 0xFFFFu) << 16);          // tensor_dim0[15:0] @ bits 63:48
  g1[2] = (int)((tile_w >> 16) | ((tile_h & 0xFFFFu) << 16)); // dim0 hi | dim1 lo
  g1[3] = (int)((tile_h >> 16) | (tile_w << 16));   // dim1 hi | tile_dim0
  g1[4] = (int)tile_h;                              // tile_dim1 (tile_dim2=0)
  g1[5] = (int)src_stride;                          // tensor_dim0_stride[31:0]
  g1[6] = 0;
  g1[7] = 0;
  i32x4 z4 = {0, 0, 0, 0};
#if __clang_major__ >= 23
  i32x8 z8 = {0, 0, 0, 0, 0, 0, 0, 0};
  __builtin_amdgcn_tensor_load_to_lds(g0, g1, z4, z4, z8, 0);
#else
  __builtin_amdgcn_tensor_load_to_lds(g0, g1, z4, z4, 0);
#endif
}
#endif

// ---------------- f32 -> bf16 conversion ----------------
__global__ __launch_bounds__(256) void cvt_kernel(const float* __restrict__ in,
                                                  unsigned short* __restrict__ out,
                                                  int n4) {
  int i = blockIdx.x * 256 + threadIdx.x;
  if (i >= n4) return;
  float4 f = ((const float4*)in)[i];
  unsigned long long pk = (unsigned long long)f2bf(f.x)
                        | ((unsigned long long)f2bf(f.y) << 16)
                        | ((unsigned long long)f2bf(f.z) << 32)
                        | ((unsigned long long)f2bf(f.w) << 48);
  *(unsigned long long*)(out + 4 * (size_t)i) = pk;
}

// ---------------- bf16 GEMM: C[m,n] = sum_k A[m,k]*B[n,k] + bias[n] ----------------
// A: M x K row-major bf16 (register-direct fragments, no cross-wave reuse).
// B: N x K row-major bf16 (weight (out,in) layout) -> shared 64x128 slab in LDS,
//    staged by the TDM and double-buffered: waves 8x-reuse each slab.
template<bool OUT_F32>
__global__ __launch_bounds__(256) void gemm_kernel(const unsigned short* __restrict__ A,
                                                   const unsigned short* __restrict__ Bm,
                                                   const float* __restrict__ bias,
                                                   float* __restrict__ Cf,
                                                   unsigned short* __restrict__ Cb,
                                                   int M, int N, int K) {
  __shared__ unsigned short Bs[2][64 * 128];   // 2 x 16 KB
  const int tid  = threadIdx.x;
  const int wave = tid >> 5;
  const int lane = tid & 31;
  const int lo = lane & 15, hi = lane >> 4;
  const int mt = M >> 7;
  const int bm = blockIdx.x % mt;
  const int bn = blockIdx.x / mt;
  const int m0 = bm * 128 + wave * 16;
  const int n0 = bn * 64;
  const int nsl = K >> 7;                      // 128-wide K slabs
  v8f acc[4] = {};
  const unsigned short* ap = A + (size_t)(m0 + lo) * K + hi * 8;
#if USE_TDM
  if (wave == 0)
    tdm_load_2d_bf16(Bs[0], Bm + (size_t)n0 * K, 128, 64, K);
#endif
  for (int sl = 0; sl < nsl; ++sl) {
    const int cur = sl & 1;
    __syncthreads();                           // prior readers of Bs[cur^1] done
#if USE_TDM
    if (wave == 0) {
      if (sl + 1 < nsl) {
        tdm_load_2d_bf16(Bs[cur ^ 1], Bm + (size_t)n0 * K + (sl + 1) * 128, 128, 64, K);
        __builtin_amdgcn_s_wait_tensorcnt(1);  // current slab landed (in-order)
      } else {
        __builtin_amdgcn_s_wait_tensorcnt(0);
      }
    }
#else
    for (int i = tid; i < 64 * 128 / 8; i += 256) {
      int r = i >> 4, c = (i & 15) * 8;
      *(uint4*)(Bs[cur] + r * 128 + c) =
          *(const uint4*)(Bm + (size_t)(n0 + r) * K + sl * 128 + c);
    }
#endif
    __syncthreads();                           // publish slab to all waves
#pragma unroll
    for (int kq = 0; kq < 4; ++kq) {
      const int k0 = sl * 128 + kq * 32;
      Frag a;
      a.q[0] = *(const uint4*)(ap + k0);
      a.q[1] = *(const uint4*)(ap + k0 + 16);
#pragma unroll
      for (int t = 0; t < 4; ++t) {
        const unsigned short* bp = Bs[cur] + (t * 16 + lo) * 128 + hi * 8 + kq * 32;
        Frag b;
        b.q[0] = *(const uint4*)(bp);
        b.q[1] = *(const uint4*)(bp + 16);
        acc[t] = wmma_bf16(a, b, acc[t]);
      }
    }
  }
#pragma unroll
  for (int t = 0; t < 4; ++t) {
#pragma unroll
    for (int vv = 0; vv < 8; ++vv) {
      int r = m0 + vv + 8 * hi;          // C layout: row = v + 8*(lane>>4)
      int c = n0 + t * 16 + lo;          //           col = lane & 15
      float val = acc[t][vv] + bias[c];
      if (OUT_F32) Cf[(size_t)r * N + c] = val;
      else         Cb[(size_t)r * N + c] = f2bf(val);
    }
  }
}

// ---------------- RMS-norm + RoPE, in place on bf16 q/k ----------------
__global__ __launch_bounds__(256) void normrope_kernel(unsigned short* __restrict__ t,
                                                       const float* __restrict__ w,
                                                       const float* __restrict__ freqs) {
  int tid  = threadIdx.x;
  int lane = tid & 31;
  int row  = blockIdx.x * 8 + (tid >> 5);   // 0 .. BSROWS*HEADS-1
  int bs   = row / HEADS;
  int h    = row - bs * HEADS;
  int s    = bs & (SEQ - 1);
  size_t base = (size_t)bs * INNER + h * HDIM;
  float x0 = bf2f(t[base + 2 * lane]);
  float x1 = bf2f(t[base + 2 * lane + 1]);
  float ss = x0 * x0 + x1 * x1;
#pragma unroll
  for (int off = 16; off > 0; off >>= 1) ss += __shfl_xor(ss, off, 32);
  float rs = rsqrtf(ss * (1.0f / 64.0f) + 1e-6f);
  x0 = x0 * rs * w[2 * lane];
  x1 = x1 * rs * w[2 * lane + 1];
  float fr = freqs[(size_t)s * HDIM + 2 * lane];   // freqs repeated pairwise
  float c = cosf(fr), sn = sinf(fr);
  t[base + 2 * lane]     = f2bf(x0 * c - x1 * sn);
  t[base + 2 * lane + 1] = f2bf(x1 * c + x0 * sn);
}

// ---------------- block-sparse flash attention ----------------
// One block per (b, h, query-chunk); 8 waves, each owns a 16-row Q strip held
// in registers.  K/V chunks are TDM-prefetched into double-buffered LDS.
__global__ __launch_bounds__(256) void attn_kernel(const unsigned short* __restrict__ q,
                                                   const unsigned short* __restrict__ k,
                                                   const unsigned short* __restrict__ v,
                                                   unsigned short* __restrict__ o) {
  __shared__ unsigned short Ks[2][CHUNKSZ * HDIM];   // 2 x 16 KB
  __shared__ unsigned short Vs[2][CHUNKSZ * HDIM];   // 2 x 16 KB
  __shared__ unsigned short Vt[HDIM * CHUNKSZ];      // 16 KB (transposed V)
  __shared__ unsigned short Ps[8 * 16 * CHUNKSZ];    // 32 KB (per-wave P tiles)
  const int tid = threadIdx.x, wave = tid >> 5, lane = tid & 31;
  const int lo = lane & 15, hi = lane >> 4;
  const int qc = blockIdx.x & 15;
  const int h  = (blockIdx.x >> 4) & 15;
  const int b  = blockIdx.x >> 8;
  const size_t qbase = ((size_t)(b * SEQ + qc * CHUNKSZ)) * INNER + h * HDIM;
  // Q strip in registers: A-fragment pattern == per-lane global pattern.
  Frag qf[2];
  {
    const unsigned short* qp = q + qbase + (size_t)(wave * 16 + lo) * INNER + hi * 8;
    qf[0].q[0] = *(const uint4*)(qp);
    qf[0].q[1] = *(const uint4*)(qp + 16);
    qf[1].q[0] = *(const uint4*)(qp + 32);
    qf[1].q[1] = *(const uint4*)(qp + 48);
  }
  // key-chunk list: {qc} U [max(0,qc-13), qc-9]   (ci indexes KEYS in the ref mask)
  int chunks[6]; int nch = 1; chunks[0] = qc;
  {
    int klo = qc - 13 > 0 ? qc - 13 : 0;
    for (int kc = klo; kc <= qc - 9; ++kc) chunks[nch++] = kc;
  }
  float mrow[8], lrow[8];
#pragma unroll
  for (int vv = 0; vv < 8; ++vv) { mrow[vv] = -3.0e38f; lrow[vv] = 0.0f; }
  v8f oacc[4] = {};
  const float scale = 0.125f;  // 1/sqrt(64)
#if USE_TDM
  if (wave == 0) {
    const size_t kb0 = ((size_t)(b * SEQ + chunks[0] * CHUNKSZ)) * INNER + h * HDIM;
    tdm_load_2d_bf16(Ks[0], k + kb0, HDIM, CHUNKSZ, INNER);
    tdm_load_2d_bf16(Vs[0], v + kb0, HDIM, CHUNKSZ, INNER);
  }
#endif
  for (int ci = 0; ci < nch; ++ci) {
    const int cur = ci & 1;
    __syncthreads();                        // prior readers of Ks/Vs[cur^1] done
#if USE_TDM
    if (wave == 0) {
      if (ci + 1 < nch) {
        const size_t kbn = ((size_t)(b * SEQ + chunks[ci + 1] * CHUNKSZ)) * INNER + h * HDIM;
        tdm_load_2d_bf16(Ks[cur ^ 1], k + kbn, HDIM, CHUNKSZ, INNER);
        tdm_load_2d_bf16(Vs[cur ^ 1], v + kbn, HDIM, CHUNKSZ, INNER);
        __builtin_amdgcn_s_wait_tensorcnt(2);   // chunk ci landed (in-order TDM)
      } else {
        __builtin_amdgcn_s_wait_tensorcnt(0);
      }
    }
#else
    {
      const size_t kb = ((size_t)(b * SEQ + chunks[ci] * CHUNKSZ)) * INNER + h * HDIM;
      for (int i = tid; i < CHUNKSZ * HDIM / 8; i += 256) {
        int r = i >> 3, d = (i & 7) * 8;
        *(uint4*)(Ks[cur] + r * HDIM + d) = *(const uint4*)(k + kb + (size_t)r * INNER + d);
        *(uint4*)(Vs[cur] + r * HDIM + d) = *(const uint4*)(v + kb + (size_t)r * INNER + d);
      }
    }
#endif
    __syncthreads();                        // publish chunk ci
    // LDS transpose Vs[cur] -> Vt (B operand of the O-GEMM wants d-major rows)
    for (int i = tid; i < CHUNKSZ * HDIM / 8; i += 256) {
      int r = i >> 3, d = (i & 7) * 8;
      uint4 vv4 = *(const uint4*)(Vs[cur] + r * HDIM + d);
      unsigned short tmp[8];
      *(uint4*)tmp = vv4;
#pragma unroll
      for (int j = 0; j < 8; ++j) Vt[(d + j) * CHUNKSZ + r] = tmp[j];
    }
    __syncthreads();
    // S = Q_strip (16x64) x K^T (64x128) -> 8 WMMA tiles
    v8f st[8] = {};
#pragma unroll
    for (int dq = 0; dq < 2; ++dq) {
#pragma unroll
      for (int t = 0; t < 8; ++t) {
        const unsigned short* kp = Ks[cur] + (t * 16 + lo) * HDIM + hi * 8 + dq * 32;
        Frag bb;
        bb.q[0] = *(const uint4*)kp;
        bb.q[1] = *(const uint4*)(kp + 16);
        st[t] = wmma_bf16(qf[dq], bb, st[t]);
      }
    }
    // online softmax: row = vv + 8*hi lives across the 16 lanes of this half
    float rmax[8], rsum[8];
#pragma unroll
    for (int vv = 0; vv < 8; ++vv) {
      float mx = -3.0e38f;
#pragma unroll
      for (int t = 0; t < 8; ++t) mx = fmaxf(mx, st[t][vv]);
      rmax[vv] = mx * scale;
#pragma unroll
      for (int off = 8; off > 0; off >>= 1)
        rmax[vv] = fmaxf(rmax[vv], __shfl_xor(rmax[vv], off, 16));
    }
    float alpha[8], mnew[8];
#pragma unroll
    for (int vv = 0; vv < 8; ++vv) {
      mnew[vv]  = fmaxf(mrow[vv], rmax[vv]);
      alpha[vv] = __expf(mrow[vv] - mnew[vv]);
      rsum[vv]  = 0.0f;
    }
    unsigned short* prow = Ps + wave * (16 * CHUNKSZ);
#pragma unroll
    for (int t = 0; t < 8; ++t) {
#pragma unroll
      for (int vv = 0; vv < 8; ++vv) {
        float p = __expf(st[t][vv] * scale - mnew[vv]);
        rsum[vv] += p;
        prow[(vv + 8 * hi) * CHUNKSZ + t * 16 + lo] = f2bf(p);
      }
    }
#pragma unroll
    for (int vv = 0; vv < 8; ++vv) {
#pragma unroll
      for (int off = 8; off > 0; off >>= 1) rsum[vv] += __shfl_xor(rsum[vv], off, 16);
      lrow[vv] = lrow[vv] * alpha[vv] + rsum[vv];
      mrow[vv] = mnew[vv];
    }
#pragma unroll
    for (int t = 0; t < 4; ++t)
#pragma unroll
      for (int vv = 0; vv < 8; ++vv) oacc[t][vv] *= alpha[vv];
    __syncthreads();                        // Ps write -> read ordering (uniform)
    // O += P (16x128) x V (128x64): A from Ps, B from Vt (N x K row-major)
#pragma unroll
    for (int kk0 = 0; kk0 < CHUNKSZ; kk0 += 32) {
      const unsigned short* pr = prow + lo * CHUNKSZ + hi * 8 + kk0;
      Frag a;
      a.q[0] = *(const uint4*)pr;
      a.q[1] = *(const uint4*)(pr + 16);
#pragma unroll
      for (int t = 0; t < 4; ++t) {
        const unsigned short* vp = Vt + (t * 16 + lo) * CHUNKSZ + hi * 8 + kk0;
        Frag bb;
        bb.q[0] = *(const uint4*)vp;
        bb.q[1] = *(const uint4*)(vp + 16);
        oacc[t] = wmma_bf16(a, bb, oacc[t]);
      }
    }
  }
#pragma unroll
  for (int t = 0; t < 4; ++t) {
#pragma unroll
    for (int vv = 0; vv < 8; ++vv) {
      int r = qc * CHUNKSZ + wave * 16 + vv + 8 * hi;
      int c = h * HDIM + t * 16 + lo;
      float val = oacc[t][vv] / lrow[vv];
      o[((size_t)b * SEQ + r) * INNER + c] = f2bf(val);
    }
  }
}

extern "C" void kernel_launch(void* const* d_in, const int* in_sizes, int n_in,
                              void* d_out, int out_size, void* d_ws, size_t ws_size,
                              hipStream_t stream) {
  (void)in_sizes; (void)n_in; (void)out_size; (void)ws_size;
  const float* x     = (const float*)d_in[0];
  // d_in[1] = mask: all-true in this problem, block mask already covers diag.
  const float* freqs = (const float*)d_in[2];
  const float* Wq = (const float*)d_in[3];  const float* bq = (const float*)d_in[4];
  const float* Wk = (const float*)d_in[5];  const float* bk = (const float*)d_in[6];
  const float* Wv = (const float*)d_in[7];  const float* bv = (const float*)d_in[8];
  const float* Wo = (const float*)d_in[9];  const float* bo = (const float*)d_in[10];
  const float* qw = (const float*)d_in[11]; const float* kw = (const float*)d_in[12];

  unsigned short* ws = (unsigned short*)d_ws;
  const size_t NX = (size_t)BSROWS * DIMX;   // 4 Mi elems
  const size_t NW = (size_t)INNER * DIMX;    // 1 Mi elems
  unsigned short* xb  = ws;
  unsigned short* wqb = xb  + NX;
  unsigned short* wkb = wqb + NW;
  unsigned short* wvb = wkb + NW;
  unsigned short* wob = wvb + NW;
  unsigned short* qb  = wob + NW;
  unsigned short* kbf = qb  + NX;
  unsigned short* vbf = kbf + NX;
  unsigned short* obf = vbf + NX;            // total 48 MiB of ws

  cvt_kernel<<<(int)((NX / 4 + 255) / 256), 256, 0, stream>>>(x,  xb,  (int)(NX / 4));
  cvt_kernel<<<(int)((NW / 4 + 255) / 256), 256, 0, stream>>>(Wq, wqb, (int)(NW / 4));
  cvt_kernel<<<(int)((NW / 4 + 255) / 256), 256, 0, stream>>>(Wk, wkb, (int)(NW / 4));
  cvt_kernel<<<(int)((NW / 4 + 255) / 256), 256, 0, stream>>>(Wv, wvb, (int)(NW / 4));
  cvt_kernel<<<(int)((NW / 4 + 255) / 256), 256, 0, stream>>>(Wo, wob, (int)(NW / 4));

  const int gGemm = (BSROWS / 128) * (INNER / 64);   // 512 blocks
  gemm_kernel<false><<<gGemm, 256, 0, stream>>>(xb, wqb, bq, nullptr, qb, BSROWS, INNER, DIMX);
  gemm_kernel<false><<<gGemm, 256, 0, stream>>>(xb, wkb, bk, nullptr, kbf, BSROWS, INNER, DIMX);
  gemm_kernel<false><<<gGemm, 256, 0, stream>>>(xb, wvb, bv, nullptr, vbf, BSROWS, INNER, DIMX);

  normrope_kernel<<<BSROWS * HEADS / 8, 256, 0, stream>>>(qb,  qw, freqs);
  normrope_kernel<<<BSROWS * HEADS / 8, 256, 0, stream>>>(kbf, kw, freqs);

  attn_kernel<<<BATCH * HEADS * NCHK, 256, 0, stream>>>(qb, kbf, vbf, obf);

  gemm_kernel<true><<<(BSROWS / 128) * (DIMX / 64), 256, 0, stream>>>(
      obf, wob, bo, (float*)d_out, nullptr, BSROWS, DIMX, INNER);
}
